// GATClassifier_34299608826091
// MI455X (gfx1250) — matrix operand order repaired and checked by reading, not compiled, not run on hardware
//
#include <hip/hip_runtime.h>
#include <hip/hip_bf16.h>
#include <stdint.h>

// ---------------------------------------------------------------------------
// Types for CDNA5 WMMA (wave32): v_wmma_f32_16x16x32_bf16
// ---------------------------------------------------------------------------
typedef __attribute__((ext_vector_type(16))) __bf16        v16bf;
typedef __attribute__((ext_vector_type(8)))  float         v8f;
typedef __attribute__((ext_vector_type(4)))  unsigned int  uvec4;
typedef __attribute__((ext_vector_type(2)))  unsigned int  uvec2;

#define NEG_SLOPE 0.2f

__device__ __forceinline__ unsigned short f2bf(float f) {
    unsigned int u = __float_as_uint(f);
    u += 0x7FFFu + ((u >> 16) & 1u);            // round-to-nearest-even
    return (unsigned short)(u >> 16);
}
__device__ __forceinline__ float bf2f(unsigned short s) {
    return __uint_as_float(((unsigned int)s) << 16);
}
__device__ __forceinline__ float pick4(float4 v, int h) {
    float r = v.x;
    if (h == 1) r = v.y;
    if (h == 2) r = v.z;
    if (h == 3) r = v.w;
    return r;
}

// ---------------------------------------------------------------------------
// 1) Brute-force kNN (k=4, loop=True) per batch, LDS tiled.
// coords: [B,P,2] f32 ; nbr: [B*P,4] global node ids
// ---------------------------------------------------------------------------
__global__ void knn_kernel(const float* __restrict__ coords,
                           int* __restrict__ nbr, int P) {
    const int tilesPerBatch = P / 256;
    const int b    = blockIdx.x / tilesPerBatch;
    const int tile = blockIdx.x % tilesPerBatch;
    const int i     = tile * 256 + threadIdx.x;
    const int gbase = b * P;

    float2 ci = ((const float2*)coords)[gbase + i];
    float bd0 = 3e38f, bd1 = 3e38f, bd2 = 3e38f, bd3 = 3e38f;
    int   bi0 = 0,     bi1 = 0,     bi2 = 0,     bi3 = 0;

    __shared__ float2 sc[256];
    for (int t = 0; t < tilesPerBatch; ++t) {
        sc[threadIdx.x] = ((const float2*)coords)[gbase + t * 256 + threadIdx.x];
        __syncthreads();
        #pragma unroll 4
        for (int jj = 0; jj < 256; ++jj) {
            float dx = ci.x - sc[jj].x;
            float dy = ci.y - sc[jj].y;
            float d  = dx * dx + dy * dy;
            int   j  = t * 256 + jj;
            if (d < bd3) {                       // strict < : ties keep lower index
                if (d < bd2) {
                    bd3 = bd2; bi3 = bi2;
                    if (d < bd1) {
                        bd2 = bd1; bi2 = bi1;
                        if (d < bd0) { bd1 = bd0; bi1 = bi0; bd0 = d; bi0 = j; }
                        else         { bd1 = d;   bi1 = j; }
                    } else { bd2 = d; bi2 = j; }
                } else { bd3 = d; bi3 = j; }
            }
        }
        __syncthreads();
    }
    const int gi = gbase + i;
    nbr[gi * 4 + 0] = gbase + bi0;
    nbr[gi * 4 + 1] = gbase + bi1;
    nbr[gi * 4 + 2] = gbase + bi2;
    nbr[gi * 4 + 3] = gbase + bi3;
}

// ---------------------------------------------------------------------------
// 2) f32 -> bf16 conversion (4 elements / thread)
// ---------------------------------------------------------------------------
__global__ void f32_to_bf16_kernel(const float* __restrict__ src,
                                   unsigned short* __restrict__ dst, int n4) {
    int i = blockIdx.x * blockDim.x + threadIdx.x;
    if (i >= n4) return;
    float4 v = ((const float4*)src)[i];
    uvec2 r;
    r.x = (unsigned)f2bf(v.x) | ((unsigned)f2bf(v.y) << 16);
    r.y = (unsigned)f2bf(v.z) | ((unsigned)f2bf(v.w) << 16);
    ((uvec2*)dst)[i] = r;
}

// ---------------------------------------------------------------------------
// 3) Pack W[256,256] (f32 row-major, K x N) into per-lane B-fragment layout:
//    wfrag[((ct*8 + kt)*32 + lane)*16 + e]
//    lane<16 : K = kt*32 +      e , N = ct*16 + lane
//    lane>=16: K = kt*32 + 16 + e , N = ct*16 + lane-16   (contiguous k, same order as A)
// ---------------------------------------------------------------------------
__global__ void pack_w_kernel(const float* __restrict__ W,
                              unsigned short* __restrict__ wfrag) {
    int t = blockIdx.x * blockDim.x + threadIdx.x;   // 16*8*32*16 = 65536
    int e    = t & 15;
    int lane = (t >> 4) & 31;
    int kt   = (t >> 9) & 7;
    int ct   = t >> 12;
    int kh   = lane >> 4;
    int nl   = lane & 15;
    int K  = kt * 32 + kh * 16 + e;
    int Nc = ct * 16 + nl;
    wfrag[t] = f2bf(W[K * 256 + Nc]);
}

// ---------------------------------------------------------------------------
// 4) WMMA GEMM: h[Nn,256] = xb[Nn,256](bf16) @ W(bf16, pre-swizzled fragments)
//    One wave -> one 16-row stripe x full 256 cols (16 accumulator tiles).
// ---------------------------------------------------------------------------
union Frag16 { v16bf v; uvec4 u[2]; };

__global__ __launch_bounds__(256) void gemm_wmma_kernel(
    const unsigned short* __restrict__ xb,     // [Nn,256] bf16 row-major
    const unsigned short* __restrict__ wfrag,  // [16][8][32][16] bf16
    float* __restrict__ hout)                  // [Nn,256] f32
{
    const int wave = threadIdx.x >> 5;
    const int lane = threadIdx.x & 31;
    const int kh   = lane >> 4;                 // K-half selector
    const int m    = lane & 15;
    const int rowBase = (blockIdx.x * 8 + wave) * 16;

    v8f acc[16];
    #pragma unroll
    for (int ct = 0; ct < 16; ++ct)
        #pragma unroll
        for (int r = 0; r < 8; ++r) acc[ct][r] = 0.0f;

    const unsigned short* arow = xb + (size_t)(rowBase + m) * 256;

    #pragma unroll
    for (int kt = 0; kt < 8; ++kt) {
        // A fragment: row m, 16 contiguous elts at K = kt*32 + 16*kh
        Frag16 a;
        const int k0 = kt * 32 + kh * 16;
        a.u[0] = *(const uvec4*)(arow + k0);
        a.u[1] = *(const uvec4*)(arow + k0 + 8);
        #pragma unroll
        for (int ct = 0; ct < 16; ++ct) {
            Frag16 b;
            const unsigned short* bp = wfrag + ((size_t)((ct * 8 + kt) * 32 + lane) * 16);
            b.u[0] = ((const uvec4*)bp)[0];
            b.u[1] = ((const uvec4*)bp)[1];
            acc[ct] = __builtin_amdgcn_wmma_f32_16x16x32_bf16(
                false, a.v, false, b.v, (short)0, acc[ct], false, false);
        }
    }

    // C/D layout: VGPR r, lanes 0-15 -> row r, lanes 16-31 -> row r+8, col = lane&15
    #pragma unroll
    for (int ct = 0; ct < 16; ++ct) {
        const int col = ct * 16 + m;
        #pragma unroll
        for (int r = 0; r < 8; ++r)
            hout[(size_t)(rowBase + kh * 8 + r) * 256 + col] = acc[ct][r];
    }
}

// ---------------------------------------------------------------------------
// 5) Attention coefficients: asrc[n,h] = <h_row[h], a_src[h]>, same for adst
// ---------------------------------------------------------------------------
__global__ void coef_kernel(const float* __restrict__ h,
                            const float* __restrict__ a_src,
                            const float* __restrict__ a_dst,
                            float* __restrict__ asrc,
                            float* __restrict__ adst, int Nn) {
    int n = blockIdx.x * blockDim.x + threadIdx.x;
    if (n >= Nn) return;
    const float* row = h + (size_t)n * 256;
    #pragma unroll
    for (int hh = 0; hh < 4; ++hh) {
        float s = 0.f, t = 0.f;
        #pragma unroll
        for (int f = 0; f < 64; f += 4) {
            float4 v  = *(const float4*)(row + hh * 64 + f);
            float4 sa = *(const float4*)(a_src + hh * 64 + f);
            float4 da = *(const float4*)(a_dst + hh * 64 + f);
            s += v.x * sa.x + v.y * sa.y + v.z * sa.z + v.w * sa.w;
            t += v.x * da.x + v.y * da.y + v.z * da.z + v.w * da.w;
        }
        asrc[n * 4 + hh] = s;
        adst[n * 4 + hh] = t;
    }
}

// ---------------------------------------------------------------------------
// 6) Attention aggregate (layers 0/1): wave per node, lane owns 8 channels.
//    softmax over k=4 per head, out = sum alpha*h_j + b, ELU, write bf16.
// ---------------------------------------------------------------------------
__global__ void attn_kernel(const float* __restrict__ h,
                            const int* __restrict__ nbr,
                            const float* __restrict__ asrc,
                            const float* __restrict__ adst,
                            const float* __restrict__ bias,
                            unsigned short* __restrict__ xnext) {
    const int wave = threadIdx.x >> 5;
    const int lane = threadIdx.x & 31;
    const int n = blockIdx.x * 8 + wave;

    int4 nb = *(const int4*)(nbr + (size_t)n * 4);
    int nj0 = nb.x, nj1 = nb.y, nj2 = nb.z, nj3 = nb.w;

    const int c0   = lane * 8;
    const int head = c0 >> 6;

    float4 ad = *(const float4*)(adst + (size_t)n * 4);
    float adh = pick4(ad, head);

    float4 a0 = *(const float4*)(asrc + (size_t)nj0 * 4);
    float4 a1 = *(const float4*)(asrc + (size_t)nj1 * 4);
    float4 a2 = *(const float4*)(asrc + (size_t)nj2 * 4);
    float4 a3 = *(const float4*)(asrc + (size_t)nj3 * 4);

    float e0 = pick4(a0, head) + adh; e0 = e0 > 0.f ? e0 : NEG_SLOPE * e0;
    float e1 = pick4(a1, head) + adh; e1 = e1 > 0.f ? e1 : NEG_SLOPE * e1;
    float e2 = pick4(a2, head) + adh; e2 = e2 > 0.f ? e2 : NEG_SLOPE * e2;
    float e3 = pick4(a3, head) + adh; e3 = e3 > 0.f ? e3 : NEG_SLOPE * e3;

    float mx = fmaxf(fmaxf(e0, e1), fmaxf(e2, e3));
    e0 = __expf(e0 - mx); e1 = __expf(e1 - mx);
    e2 = __expf(e2 - mx); e3 = __expf(e3 - mx);
    float inv = 1.f / (e0 + e1 + e2 + e3);
    e0 *= inv; e1 *= inv; e2 *= inv; e3 *= inv;

    float o[8];
    #pragma unroll
    for (int f = 0; f < 8; ++f) o[f] = 0.f;

    const int njs[4] = {nj0, nj1, nj2, nj3};
    const float ws[4] = {e0, e1, e2, e3};
    #pragma unroll
    for (int k = 0; k < 4; ++k) {
        const float4* hp = (const float4*)(h + (size_t)njs[k] * 256 + c0);
        float4 x0 = hp[0], x1 = hp[1];
        float w = ws[k];
        o[0] += w * x0.x; o[1] += w * x0.y; o[2] += w * x0.z; o[3] += w * x0.w;
        o[4] += w * x1.x; o[5] += w * x1.y; o[6] += w * x1.z; o[7] += w * x1.w;
    }

    unsigned short* outp = xnext + (size_t)n * 256 + c0;
    #pragma unroll
    for (int f = 0; f < 8; ++f) {
        float r = o[f] + bias[c0 + f];
        r = r > 0.f ? r : (__expf(r) - 1.f);   // ELU
        outp[f] = f2bf(r);
    }
}

// ---------------------------------------------------------------------------
// 7) Layer-2 tiny GEMM (256->2) fused with its attention coefficients.
// ---------------------------------------------------------------------------
__global__ void gemm2_coef_kernel(const unsigned short* __restrict__ x2b,
                                  const float* __restrict__ W2,   // [256,2]
                                  const float* __restrict__ av,   // [2]
                                  const float* __restrict__ dv,   // [2]
                                  float* __restrict__ y2,
                                  float* __restrict__ as2,
                                  float* __restrict__ ad2, int Nn) {
    int n = blockIdx.x * blockDim.x + threadIdx.x;
    if (n >= Nn) return;
    const uvec4* xr = (const uvec4*)(x2b + (size_t)n * 256);
    float acc0 = 0.f, acc1 = 0.f;
    for (int c8 = 0; c8 < 32; ++c8) {
        uvec4 u = xr[c8];
        int base = c8 * 8;
        #pragma unroll
        for (int q = 0; q < 4; ++q) {
            unsigned w = u[q];
            float vlo = bf2f((unsigned short)(w & 0xFFFFu));
            float vhi = bf2f((unsigned short)(w >> 16));
            int c = base + q * 2;
            acc0 += vlo * W2[c * 2 + 0] + vhi * W2[(c + 1) * 2 + 0];
            acc1 += vlo * W2[c * 2 + 1] + vhi * W2[(c + 1) * 2 + 1];
        }
    }
    y2[(size_t)n * 2 + 0] = acc0;
    y2[(size_t)n * 2 + 1] = acc1;
    as2[n] = acc0 * av[0] + acc1 * av[1];
    ad2[n] = acc0 * dv[0] + acc1 * dv[1];
}

// ---------------------------------------------------------------------------
// 8) Final attention (H=1, F=2), writes d_out [B,P,2] f32.
// ---------------------------------------------------------------------------
__global__ void attn_final_kernel(const float* __restrict__ y2,
                                  const int* __restrict__ nbr,
                                  const float* __restrict__ as2,
                                  const float* __restrict__ ad2,
                                  const float* __restrict__ b2,
                                  float* __restrict__ out, int Nn) {
    int n = blockIdx.x * blockDim.x + threadIdx.x;
    if (n >= Nn) return;
    int4 nb = *(const int4*)(nbr + (size_t)n * 4);
    const int njs[4] = {nb.x, nb.y, nb.z, nb.w};
    float ad = ad2[n];
    float e[4];
    float mx = -3e38f;
    #pragma unroll
    for (int k = 0; k < 4; ++k) {
        float v = as2[njs[k]] + ad;
        v = v > 0.f ? v : NEG_SLOPE * v;
        e[k] = v; mx = fmaxf(mx, v);
    }
    float s = 0.f;
    #pragma unroll
    for (int k = 0; k < 4; ++k) { e[k] = __expf(e[k] - mx); s += e[k]; }
    float inv = 1.f / s;
    float o0 = 0.f, o1 = 0.f;
    #pragma unroll
    for (int k = 0; k < 4; ++k) {
        float w = e[k] * inv;
        o0 += w * y2[(size_t)njs[k] * 2 + 0];
        o1 += w * y2[(size_t)njs[k] * 2 + 1];
    }
    out[(size_t)n * 2 + 0] = o0 + b2[0];
    out[(size_t)n * 2 + 1] = o1 + b2[1];
}

// ---------------------------------------------------------------------------
// Host launcher
// ---------------------------------------------------------------------------
extern "C" void kernel_launch(void* const* d_in, const int* in_sizes, int n_in,
                              void* d_out, int out_size, void* d_ws, size_t ws_size,
                              hipStream_t stream) {
    const float* coords = (const float*)d_in[0];
    const float* x      = (const float*)d_in[1];
    const float* W0   = (const float*)d_in[2];
    const float* as0  = (const float*)d_in[3];
    const float* ad0  = (const float*)d_in[4];
    const float* b0   = (const float*)d_in[5];
    const float* W1   = (const float*)d_in[6];
    const float* as1  = (const float*)d_in[7];
    const float* ad1  = (const float*)d_in[8];
    const float* b1   = (const float*)d_in[9];
    const float* W2   = (const float*)d_in[10];
    const float* as2v = (const float*)d_in[11];
    const float* ad2v = (const float*)d_in[12];
    const float* b2   = (const float*)d_in[13];
    (void)n_in; (void)out_size; (void)ws_size;

    const int Nn = in_sizes[1] / 256;     // 131072 nodes
    const int P  = 8192;                  // points per image
    const int BP = in_sizes[0] / 2;       // B*P

    // Workspace carve-up (all 256B aligned)
    char* ws = (char*)d_ws;
    size_t off = 0;
    auto alloc = [&](size_t bytes) -> void* {
        void* p = ws + off;
        off += (bytes + 255) & ~(size_t)255;
        return p;
    };
    int*            nbr   = (int*)           alloc((size_t)Nn * 4 * sizeof(int));
    unsigned short* xb    = (unsigned short*)alloc((size_t)Nn * 256 * 2);
    float*          hbuf  = (float*)         alloc((size_t)Nn * 256 * 4);
    float*          asrc  = (float*)         alloc((size_t)Nn * 4 * 4);
    float*          adst  = (float*)         alloc((size_t)Nn * 4 * 4);
    unsigned short* wfrag = (unsigned short*)alloc(256 * 256 * 2);
    float*          y2    = (float*)         alloc((size_t)Nn * 2 * 4);
    float*          asc2  = (float*)         alloc((size_t)Nn * 4);
    float*          adc2  = (float*)         alloc((size_t)Nn * 4);

    // kNN graph
    knn_kernel<<<BP / 256, 256, 0, stream>>>(coords, nbr, P);
    // bf16 copy of layer-0 input
    f32_to_bf16_kernel<<<(Nn * 64 + 255) / 256, 256, 0, stream>>>(x, xb, Nn * 64);

    // ---- layer 0 ----
    pack_w_kernel<<<65536 / 256, 256, 0, stream>>>(W0, wfrag);
    gemm_wmma_kernel<<<Nn / 128, 256, 0, stream>>>(xb, wfrag, hbuf);
    coef_kernel<<<(Nn + 255) / 256, 256, 0, stream>>>(hbuf, as0, ad0, asrc, adst, Nn);
    attn_kernel<<<Nn / 8, 256, 0, stream>>>(hbuf, nbr, asrc, adst, b0, xb);

    // ---- layer 1 ----
    pack_w_kernel<<<65536 / 256, 256, 0, stream>>>(W1, wfrag);
    gemm_wmma_kernel<<<Nn / 128, 256, 0, stream>>>(xb, wfrag, hbuf);
    coef_kernel<<<(Nn + 255) / 256, 256, 0, stream>>>(hbuf, as1, ad1, asrc, adst, Nn);
    attn_kernel<<<Nn / 8, 256, 0, stream>>>(hbuf, nbr, asrc, adst, b1, xb);

    // ---- layer 2 (256 -> 2) + final attention ----
    gemm2_coef_kernel<<<(Nn + 255) / 256, 256, 0, stream>>>(xb, W2, as2v, ad2v,
                                                            y2, asc2, adc2, Nn);
    attn_final_kernel<<<(Nn + 255) / 256, 256, 0, stream>>>(y2, nbr, asc2, adc2,
                                                            b2, (float*)d_out, Nn);
}
